// GCViTBlock_34273839022292
// MI455X (gfx1250) — matrix operand order repaired and checked by reading, not compiled, not run on hardware
//
#include <hip/hip_runtime.h>
#include <hip/hip_bf16.h>

// ---------------------------------------------------------------------------
// GCViT block for MI455X (gfx1250, wave32, WMMA bf16 w/ f32 accumulation)
// ---------------------------------------------------------------------------

typedef __bf16 bf16_t;
typedef __attribute__((ext_vector_type(16))) __bf16 v16bf;
typedef __attribute__((ext_vector_type(8)))  float  v8f;

// Native hardware convert (RNE) instead of software bit-twiddled rounding:
// lets the backend pick v_cvt_*bf16* instead of ~6 VALU ops per element.
__device__ __forceinline__ unsigned short f2bf(float f) {
  return __builtin_bit_cast(unsigned short, (bf16_t)f);
}
__device__ __forceinline__ bf16_t us2bf(unsigned short h) {
  return __builtin_bit_cast(bf16_t, h);
}
__device__ __forceinline__ v8f v8f_zero() {
  v8f z;
#pragma unroll
  for (int i = 0; i < 8; ++i) z[i] = 0.f;
  return z;
}

// A fragment: 16(M) x 32(K) bf16 tile from row-major storage (ld in u16 elems).
// ISA 7.12.2: lanes 0-15 & 16-31 both hold M=lane&15; lane-half offsets K by 8
// within each 16-wide K group (VGPR0..3: K 0..7 / 8..15; VGPR4..7: 16..23/24..31).
__device__ __forceinline__ v16bf load_A_frag(const unsigned short* base, int ld, int lane) {
  const int m = lane & 15, half = lane >> 4;
  const unsigned short* p = base + (size_t)m * ld;
  v16bf a;
#pragma unroll
  for (int e = 0; e < 8; ++e) a[e]     = us2bf(p[half * 8 + e]);
#pragma unroll
  for (int e = 0; e < 8; ++e) a[8 + e] = us2bf(p[16 + half * 8 + e]);
  return a;
}

// B fragment where logical B[k][n] = W[n][k], W row-major N x K (weights).
// Lane holds column n = lane&15; lane-half offsets K by 8 within each 16-wide K group (same as A).
__device__ __forceinline__ v16bf load_B_nk(const unsigned short* w, int ld, int lane) {
  const int n = lane & 15, half = lane >> 4;
  const unsigned short* p = w + (size_t)n * ld + half * 8;
  v16bf b;
#pragma unroll
  for (int e = 0; e < 16; ++e) b[e] = us2bf(p[e + (e >> 3) * 8]);
  return b;
}

// B fragment where logical B[k][n] = M[k][n], M row-major K x N.
__device__ __forceinline__ v16bf load_B_kn(const unsigned short* mkn, int ld, int lane) {
  const int n = lane & 15, half = lane >> 4;
  const unsigned short* p = mkn + (size_t)(half * 8) * ld + n;
  v16bf b;
#pragma unroll
  for (int e = 0; e < 16; ++e) b[e] = us2bf(p[(size_t)(e + (e >> 3) * 8) * ld]);
  return b;
}

__device__ __forceinline__ v8f wmma_bf16(v16bf a, v16bf b, v8f c) {
  return __builtin_amdgcn_wmma_f32_16x16x32_bf16(false, a, false, b, (short)0, c,
                                                 false, false);
}

// ---------------------------------------------------------------------------
// Kernel 0: fp32 -> bf16 weight conversion (weights then live in the 192MB L2)
// ---------------------------------------------------------------------------
__global__ void cvt_bf16_kernel(const float* __restrict__ s,
                                unsigned short* __restrict__ d, int n) {
  int i = blockIdx.x * blockDim.x + threadIdx.x;
  if (i < n) d[i] = f2bf(s[i]);
}

// ---------------------------------------------------------------------------
// Kernel 1: fused attention branch. One 8-wave workgroup per window (1024).
// LDS: XN 64x520 u16 (tokens, reused as attn output) | KV 64x1040 | P 8x64x72
// ---------------------------------------------------------------------------
__global__ __launch_bounds__(256) void gcvit_attn_kernel(
    const float* __restrict__ x, const float* __restrict__ qg,
    const float* __restrict__ g1, const float* __restrict__ b1,
    const unsigned short* __restrict__ wqkv, const float* __restrict__ qkvb,
    const float* __restrict__ relb,
    const unsigned short* __restrict__ wproj, const float* __restrict__ projb,
    float* __restrict__ xmid)
{
  extern __shared__ unsigned short smem[];
  unsigned short* XN = smem;                 // 64 x 520  (LN1 tokens / attn out)
  unsigned short* KV = smem + 64 * 520;      // 64 x 1040 (K cols 0..511, V 512..1023)
  unsigned short* PS = KV + 64 * 1040;       // 8 waves x 64 x 72 softmax probs

  const int win  = blockIdx.x;
  const int bimg = win >> 6;
  const int widx = win & 63;
  const int h1 = widx >> 3, w1 = widx & 7;
  const int tid = threadIdx.x, lane = tid & 31, wv = tid >> 5;
  const int half = lane >> 4, nl = lane & 15;

  // ---- Stage 1: LN1 over the 49 window tokens -> bf16 tokens in XN ----
  for (int n = wv; n < 64; n += 8) {
    if (n < 49) {
      const int p = h1 * 392 + (n / 7) * 56 + w1 * 7 + (n % 7);
      const float* row = x + ((size_t)bimg * 3136 + p) * 512;
      float v[16]; float s = 0.f, s2 = 0.f;
#pragma unroll
      for (int e = 0; e < 16; ++e) { float t = row[lane * 16 + e]; v[e] = t; s += t; s2 += t * t; }
      for (int m = 16; m > 0; m >>= 1) { s += __shfl_xor(s, m); s2 += __shfl_xor(s2, m); }
      const float mean = s * (1.f / 512.f);
      const float var  = s2 * (1.f / 512.f) - mean * mean;
      const float inv  = rsqrtf(var + 1e-5f);
#pragma unroll
      for (int e = 0; e < 16; ++e) {
        const int c = lane * 16 + e;
        XN[n * 520 + c] = f2bf((v[e] - mean) * inv * g1[c] + b1[c]);
      }
    } else {
#pragma unroll
      for (int e = 0; e < 16; ++e) XN[n * 520 + lane * 16 + e] = 0;  // zero pad rows
    }
  }
  __syncthreads();

  // ---- Stage 2: KV GEMM  kv(64x1024) = XN(64x512) @ Wqkv^T, 256 tiles ----
  for (int t = wv; t < 256; t += 8) {
    const int mt = t >> 6, nt = t & 63;
    v8f acc = v8f_zero();
    for (int kc = 0; kc < 16; ++kc) {
      __builtin_prefetch(wqkv + (size_t)(nt * 16) * 512 + (kc + 1) * 32, 0, 0);
      v16bf a  = load_A_frag(XN + mt * 16 * 520 + kc * 32, 520, lane);
      v16bf bb = load_B_nk(wqkv + (size_t)(nt * 16) * 512 + kc * 32, 512, lane);
      acc = wmma_bf16(a, bb, acc);
    }
#pragma unroll
    for (int r = 0; r < 8; ++r) {
      const int m = mt * 16 + r + 8 * half;
      const int n = nt * 16 + nl;
      KV[m * 1040 + n] = f2bf(acc[r] + qkvb[n]);
    }
  }
  __syncthreads();

  // ---- Stage 3: attention. Wave wv handles heads wv and wv+8 ----
  unsigned short* P = PS + wv * (64 * 72);
  for (int hh = 0; hh < 2; ++hh) {
    const int h = wv + hh * 8;
    v8f S[4][4];
#pragma unroll
    for (int mt = 0; mt < 4; ++mt) {
      // global-query A fragment (pre-scaled, bf16)
      v16bf a;
      {
        const int m = mt * 16 + nl;
        if (m < 49) {
          const float* qp = qg + (((size_t)bimg * 16 + h) * 49 + m) * 32;
#pragma unroll
          for (int e = 0; e < 8; ++e)
            a[e] = (bf16_t)(qp[half * 8 + e] * 0.17677669529663687f);
#pragma unroll
          for (int e = 0; e < 8; ++e)
            a[8 + e] = (bf16_t)(qp[16 + half * 8 + e] * 0.17677669529663687f);
        } else {
#pragma unroll
          for (int e = 0; e < 16; ++e) a[e] = us2bf((unsigned short)0);
        }
      }
#pragma unroll
      for (int nt = 0; nt < 4; ++nt) {
        v16bf bb = load_B_nk(KV + nt * 16 * 1040 + h * 32, 1040, lane);  // K matrix
        S[mt][nt] = wmma_bf16(a, bb, v8f_zero());
      }
    }
    // rel-pos bias + pad mask + softmax, done in C-fragment register layout
#pragma unroll
    for (int mt = 0; mt < 4; ++mt) {
#pragma unroll
      for (int r = 0; r < 8; ++r) {
        const int mq = mt * 16 + r + 8 * half;
        const int iq = mq / 7, jq = mq % 7;
        float mx = -3.0e38f;
#pragma unroll
        for (int nt = 0; nt < 4; ++nt) {
          const int mk = nt * 16 + nl;
          float sc = S[mt][nt][r];
          if (mk < 49) {
            if (mq < 49)
              sc += relb[((iq - mk / 7 + 6) * 13 + (jq - mk % 7 + 6)) * 16 + h];
          } else {
            sc = -3.0e38f;
          }
          S[mt][nt][r] = sc;
          mx = fmaxf(mx, sc);
        }
        for (int msk = 8; msk > 0; msk >>= 1) mx = fmaxf(mx, __shfl_xor(mx, msk));
        float sum = 0.f;
#pragma unroll
        for (int nt = 0; nt < 4; ++nt) {
          float pr = __expf(S[mt][nt][r] - mx);
          S[mt][nt][r] = pr; sum += pr;
        }
        for (int msk = 8; msk > 0; msk >>= 1) sum += __shfl_xor(sum, msk);
        const float inv = 1.f / sum;
#pragma unroll
        for (int nt = 0; nt < 4; ++nt)
          P[mq * 72 + nt * 16 + nl] = f2bf(S[mt][nt][r] * inv);
      }
    }
    // out(64x32) = P(64x64) @ V(64x32); write cols h*32..h*32+31 of XN
#pragma unroll
    for (int mt = 0; mt < 4; ++mt) {
#pragma unroll
      for (int dt = 0; dt < 2; ++dt) {
        v8f acc = v8f_zero();
#pragma unroll
        for (int kc = 0; kc < 2; ++kc) {
          v16bf a  = load_A_frag(P + mt * 16 * 72 + kc * 32, 72, lane);
          v16bf bb = load_B_kn(KV + (size_t)(kc * 32) * 1040 + 512 + h * 32 + dt * 16,
                               1040, lane);
          acc = wmma_bf16(a, bb, acc);
        }
#pragma unroll
        for (int r = 0; r < 8; ++r)
          XN[(mt * 16 + r + 8 * half) * 520 + h * 32 + dt * 16 + nl] = f2bf(acc[r]);
      }
    }
  }
  __syncthreads();

  // ---- Stage 4: proj + residual scatter (window-reverse == same row perm) ----
  for (int t = wv; t < 128; t += 8) {
    const int mt = t >> 5, nt = t & 31;
    v8f acc = v8f_zero();
    for (int kc = 0; kc < 16; ++kc) {
      __builtin_prefetch(wproj + (size_t)(nt * 16) * 512 + (kc + 1) * 32, 0, 0);
      v16bf a  = load_A_frag(XN + mt * 16 * 520 + kc * 32, 520, lane);
      v16bf bb = load_B_nk(wproj + (size_t)(nt * 16) * 512 + kc * 32, 512, lane);
      acc = wmma_bf16(a, bb, acc);
    }
#pragma unroll
    for (int r = 0; r < 8; ++r) {
      const int m = mt * 16 + r + 8 * half;
      if (m < 49) {
        const int c = nt * 16 + nl;
        const int p = h1 * 392 + (m / 7) * 56 + w1 * 7 + (m % 7);
        const size_t off = ((size_t)bimg * 3136 + p) * 512 + c;
        xmid[off] = x[off] + acc[r] + projb[c];
      }
    }
  }
}

// ---------------------------------------------------------------------------
// Kernel 2: fused MLP. One 8-wave workgroup per 32 flat rows (1568 blocks).
// Hidden 32x2048 tile lives only in LDS (saves ~410MB of HBM traffic).
// ---------------------------------------------------------------------------
__global__ __launch_bounds__(256) void gcvit_mlp_kernel(
    const float* __restrict__ xmid,
    const float* __restrict__ g2, const float* __restrict__ b2,
    const unsigned short* __restrict__ wfc1, const float* __restrict__ fc1b,
    const unsigned short* __restrict__ wfc2, const float* __restrict__ fc2b,
    float* __restrict__ out)
{
  extern __shared__ unsigned short smem[];
  unsigned short* XN = smem;              // 32 x 520  (LN2 tokens, bf16)
  unsigned short* HB = smem + 32 * 520;   // 32 x 2056 (gelu hidden, bf16)

  const size_t rowBase = (size_t)blockIdx.x * 32;
  const int tid = threadIdx.x, lane = tid & 31, wv = tid >> 5;
  const int half = lane >> 4, nl = lane & 15;

  // ---- LN2 ----
  for (int rr = wv; rr < 32; rr += 8) {
    const float* row = xmid + (rowBase + rr) * 512;
    float v[16]; float s = 0.f, s2 = 0.f;
#pragma unroll
    for (int e = 0; e < 16; ++e) { float t = row[lane * 16 + e]; v[e] = t; s += t; s2 += t * t; }
    for (int m = 16; m > 0; m >>= 1) { s += __shfl_xor(s, m); s2 += __shfl_xor(s2, m); }
    const float mean = s * (1.f / 512.f);
    const float var  = s2 * (1.f / 512.f) - mean * mean;
    const float inv  = rsqrtf(var + 1e-5f);
#pragma unroll
    for (int e = 0; e < 16; ++e) {
      const int c = lane * 16 + e;
      XN[rr * 520 + c] = f2bf((v[e] - mean) * inv * g2[c] + b2[c]);
    }
  }
  __syncthreads();

  // ---- fc1 + exact GELU: h(32x2048), 256 tiles across 8 waves ----
  for (int t = wv; t < 256; t += 8) {
    const int mt = t >> 7, nt = t & 127;
    v8f acc = v8f_zero();
    for (int kc = 0; kc < 16; ++kc) {
      __builtin_prefetch(wfc1 + (size_t)(nt * 16) * 512 + (kc + 1) * 32, 0, 0);
      v16bf a  = load_A_frag(XN + mt * 16 * 520 + kc * 32, 520, lane);
      v16bf bb = load_B_nk(wfc1 + (size_t)(nt * 16) * 512 + kc * 32, 512, lane);
      acc = wmma_bf16(a, bb, acc);
    }
#pragma unroll
    for (int r = 0; r < 8; ++r) {
      const int n = nt * 16 + nl;
      const float z = acc[r] + fc1b[n];
      const float ge = 0.5f * z * (1.f + erff(z * 0.70710678118654752f));
      HB[(mt * 16 + r + 8 * half) * 2056 + n] = f2bf(ge);
    }
  }
  __syncthreads();

  // ---- fc2 + residual -> fresh write to d_out (deterministic on replay) ----
  for (int t = wv; t < 64; t += 8) {
    const int mt = t >> 5, nt = t & 31;
    v8f acc = v8f_zero();
    for (int kc = 0; kc < 64; ++kc) {
      __builtin_prefetch(wfc2 + (size_t)(nt * 16) * 2048 + (kc + 1) * 32, 0, 0);
      v16bf a  = load_A_frag(HB + mt * 16 * 2056 + kc * 32, 2056, lane);
      v16bf bb = load_B_nk(wfc2 + (size_t)(nt * 16) * 2048 + kc * 32, 2048, lane);
      acc = wmma_bf16(a, bb, acc);
    }
#pragma unroll
    for (int r = 0; r < 8; ++r) {
      const int m = mt * 16 + r + 8 * half;
      const int c = nt * 16 + nl;
      const size_t off = (rowBase + m) * 512 + c;
      out[off] = xmid[off] + acc[r] + fc2b[c];
    }
  }
}

// ---------------------------------------------------------------------------
// Host launcher
// ---------------------------------------------------------------------------
extern "C" void kernel_launch(void* const* d_in, const int* in_sizes, int n_in,
                              void* d_out, int out_size, void* d_ws, size_t ws_size,
                              hipStream_t stream) {
  (void)in_sizes; (void)n_in; (void)out_size; (void)ws_size;

  const float* x      = (const float*)d_in[0];   // (16,56,56,512)
  const float* qg     = (const float*)d_in[1];   // (16,1,16,49,32)
  const float* n1g    = (const float*)d_in[2];
  const float* n1b    = (const float*)d_in[3];
  const float* qkv_w  = (const float*)d_in[4];   // (1024,512)
  const float* qkv_b  = (const float*)d_in[5];
  const float* relb   = (const float*)d_in[6];   // (169,16)
  const float* proj_w = (const float*)d_in[7];   // (512,512)
  const float* proj_b = (const float*)d_in[8];
  const float* n2g    = (const float*)d_in[9];
  const float* n2b    = (const float*)d_in[10];
  const float* fc1_w  = (const float*)d_in[11];  // (2048,512)
  const float* fc1_b  = (const float*)d_in[12];
  const float* fc2_w  = (const float*)d_in[13];  // (512,2048)
  const float* fc2_b  = (const float*)d_in[14];

  // workspace layout (bytes)
  char* ws = (char*)d_ws;
  constexpr size_t W_QKV_OFF  = 0;                         // 1024*512*2
  constexpr size_t W_PROJ_OFF = W_QKV_OFF  + 1048576;      // 512*512*2
  constexpr size_t W_FC1_OFF  = W_PROJ_OFF + 524288;       // 2048*512*2
  constexpr size_t W_FC2_OFF  = W_FC1_OFF  + 2097152;      // 512*2048*2
  constexpr size_t XMID_OFF   = W_FC2_OFF  + 2097152;      // 50176*512*4
  unsigned short* wqkv  = (unsigned short*)(ws + W_QKV_OFF);
  unsigned short* wproj = (unsigned short*)(ws + W_PROJ_OFF);
  unsigned short* wfc1  = (unsigned short*)(ws + W_FC1_OFF);
  unsigned short* wfc2  = (unsigned short*)(ws + W_FC2_OFF);
  float*          xmid  = (float*)(ws + XMID_OFF);

  cvt_bf16_kernel<<<(524288  + 255) / 256, 256, 0, stream>>>(qkv_w,  wqkv,  524288);
  cvt_bf16_kernel<<<(262144  + 255) / 256, 256, 0, stream>>>(proj_w, wproj, 262144);
  cvt_bf16_kernel<<<(1048576 + 255) / 256, 256, 0, stream>>>(fc1_w,  wfc1,  1048576);
  cvt_bf16_kernel<<<(1048576 + 255) / 256, 256, 0, stream>>>(fc2_w,  wfc2,  1048576);

  const size_t smemAttn = (size_t)(64 * 520 + 64 * 1040 + 8 * 64 * 72) * 2; // 273,408 B
  gcvit_attn_kernel<<<1024, 256, smemAttn, stream>>>(
      x, qg, n1g, n1b, wqkv, qkv_b, relb, wproj, proj_b, xmid);

  const size_t smemMlp = (size_t)(32 * 520 + 32 * 2056) * 2;                // 164,864 B
  gcvit_mlp_kernel<<<1568, 256, smemMlp, stream>>>(
      xmid, n2g, n2b, wfc1, fc1_b, wfc2, fc2_b, (float*)d_out);
}
